// PesudoLabelGenerator_v9_8_rev_t_3702261809694
// MI455X (gfx1250) — compile-verified, matrix operand order checked by reading, not run on hardware
//
#include <hip/hip_runtime.h>
#include <hip/hip_bf16.h>
#include <stdint.h>

namespace {

constexpr int Nn   = 6000;
constexpr int Mm   = 10000;
constexpr int Cc   = 96;
constexpr int OUTC = 20;
constexpr int LGc  = 2000;
constexpr int NTN  = 94;            // ceil(N/64)
constexpr int NTM  = 157;           // ceil(M/64)
constexpr int NP   = NTN * 64;      // 6016 padded rows
constexpr int MP   = NTM * 64;      // 10048 padded rows

typedef __attribute__((ext_vector_type(16))) _Float16 v16h;
typedef __attribute__((ext_vector_type(8)))  float    v8f;

union Frag { v16h v; uint4 q[2]; };

__device__ __forceinline__ float fast_rcp(float x) {
#if __has_builtin(__builtin_amdgcn_rcpf)
  return __builtin_amdgcn_rcpf(x);      // v_rcp_f32
#else
  return 1.0f / x;
#endif
}

// 16-bit WMMA A/B fragment layout (gfx1250, wave32):
// lane l: row/col = l&15, hi = l>>4 selects k-halves {0..7,16..23} vs {8..15,24..31}
__device__ __forceinline__ void load_frag(const _Float16* rowp, int hi, int kc, Frag& f) {
  const _Float16* p = rowp + kc + 8 * hi;
  f.q[0] = *reinterpret_cast<const uint4*>(p);        // k = kc+8hi .. +7
  f.q[1] = *reinterpret_cast<const uint4*>(p + 16);   // k = kc+16+8hi .. +7
}

// Stage a 64x96-half (12KB) panel global->LDS. Prefer the gfx1250 async
// LDS-DMA path when the toolchain exposes it; fall back to plain loads.
#if __has_builtin(__builtin_amdgcn_global_load_async_to_lds_b128) && \
    __has_builtin(__builtin_amdgcn_s_wait_asynccnt)
#define ATHENA_ASYNC_STAGE 1
#else
#define ATHENA_ASYNC_STAGE 0
#endif

#if ATHENA_ASYNC_STAGE
typedef int athena_v4i __attribute__((vector_size(16)));
typedef __attribute__((address_space(1))) athena_v4i athena_gv4i;
typedef __attribute__((address_space(3))) athena_v4i athena_lv4i;
#endif

__device__ __forceinline__ void stage_panel(const _Float16* __restrict__ gsrc,
                                            _Float16* lpanel, int t) {
#if ATHENA_ASYNC_STAGE
  athena_gv4i* g = (athena_gv4i*)(_Float16*)gsrc;   // 16B-vector elements
  athena_lv4i* l = (athena_lv4i*)lpanel;
#pragma unroll
  for (int i = 0; i < 3; ++i)   // 3 x 256 threads x 16B = 12KB
    __builtin_amdgcn_global_load_async_to_lds_b128(g + (t + i * 256),
                                                   l + (t + i * 256), 0, 0);
  __builtin_amdgcn_s_wait_asynccnt(0);
#else
  const uint* src = reinterpret_cast<const uint*>(gsrc);
  uint* dst = reinterpret_cast<uint*>(lpanel);
  for (int i = t; i < 64 * (Cc / 2); i += 256) dst[i] = src[i];
#endif
}

// reduce across the 16 lanes sharing one matrix row (lanes 0-15 / 16-31)
__device__ __forceinline__ float reduce16(float s) {
  s += __shfl_xor(s, 1, 16);
  s += __shfl_xor(s, 2, 16);
  s += __shfl_xor(s, 4, 16);
  s += __shfl_xor(s, 8, 16);
  return s;
}

// ---------------- small prep kernels -------------------------------------

__global__ void k_align(const float* __restrict__ posses, float* __restrict__ diff) {
  if (threadIdx.x != 0 || blockIdx.x != 0) return;
  float A[4][8];
  for (int i = 0; i < 4; ++i)
    for (int j = 0; j < 4; ++j) {
      A[i][j]     = posses[16 + i * 4 + j];   // posses[1]
      A[i][4 + j] = (i == j) ? 1.f : 0.f;
    }
  for (int c = 0; c < 4; ++c) {
    int p = c; float bv = fabsf(A[c][c]);
    for (int r = c + 1; r < 4; ++r) { float v = fabsf(A[r][c]); if (v > bv) { bv = v; p = r; } }
    if (p != c) for (int j = 0; j < 8; ++j) { float t = A[c][j]; A[c][j] = A[p][j]; A[p][j] = t; }
    float inv = 1.f / A[c][c];
    for (int j = 0; j < 8; ++j) A[c][j] *= inv;
    for (int r = 0; r < 4; ++r) if (r != c) {
      float fct = A[r][c];
      for (int j = 0; j < 8; ++j) A[r][j] -= fct * A[c][j];
    }
  }
  for (int i = 0; i < 4; ++i)
    for (int j = 0; j < 4; ++j) {
      float s = 0.f;
      for (int k = 0; k < 4; ++k) s += A[i][4 + k] * posses[k * 4 + j]; // inv(p1) @ p0
      diff[i * 4 + j] = s;
    }
}

__global__ void k_transform(const float* __restrict__ coords, const float* __restrict__ diff,
                            float* __restrict__ alg, float* __restrict__ cm2g) {
  int m = blockIdx.x * blockDim.x + threadIdx.x;
  if (m >= Mm) return;
  float h[4] = { coords[m * 3], coords[m * 3 + 1], coords[m * 3 + 2], 1.f };
  float s2 = 0.f;
  for (int i = 0; i < 3; ++i) {
    float v = diff[i * 4] * h[0] + diff[i * 4 + 1] * h[1] + diff[i * 4 + 2] * h[2] + diff[i * 4 + 3];
    alg[m * 3 + i] = v;
    s2 += v * v;
  }
  cm2g[m] = s2;
}

__global__ void k_prep_mean(const float* __restrict__ feat, const float* __restrict__ ori,
                            _Float16* __restrict__ f16, float* __restrict__ w1g,
                            float* __restrict__ n2og) {
  int n = blockIdx.x * blockDim.x + threadIdx.x;
  if (n >= Nn) return;
  float ss = 0.f;
  for (int k = 0; k < Cc; ++k) {
    float v = feat[n * Cc + k];
    f16[(size_t)n * Cc + k] = (_Float16)v;
    ss += v * v;
  }
  w1g[n] = sqrtf(ss);
  float x = ori[n * 3], y = ori[n * 3 + 1], z = ori[n * 3 + 2];
  n2og[n] = x * x + y * y + z * z;
}

__global__ void k_prep_sur(const float* __restrict__ feat, _Float16* __restrict__ f16,
                           float* __restrict__ w2g) {
  int m = blockIdx.x * blockDim.x + threadIdx.x;
  if (m >= Mm) return;
  float ss = 0.f;
  for (int k = 0; k < Cc; ++k) {
    float v = feat[m * Cc + k];
    f16[(size_t)m * Cc + k] = (_Float16)v;
    ss += v * v;
  }
  w2g[m] = sqrtf(ss);
}

__device__ __forceinline__ unsigned tf_rand(unsigned m) { // threefry2x32-20, key(42)
  unsigned k0 = 0u, k1 = 42u;
  unsigned ks2 = k0 ^ k1 ^ 0x1BD11BDAu;
  unsigned x0 = m + k0, x1 = k1;
  const int R0[4] = {13, 15, 26, 6}, R1[4] = {17, 29, 16, 24};
  unsigned ks[3] = {k0, k1, ks2};
  for (int i = 0; i < 5; ++i) {
    const int* R = (i & 1) ? R1 : R0;
    for (int r = 0; r < 4; ++r) { x0 += x1; x1 = (x1 << R[r]) | (x1 >> (32 - R[r])); x1 ^= x0; }
    x0 += ks[(i + 1) % 3];
    x1 += ks[(i + 2) % 3] + (unsigned)(i + 1);
  }
  return x0;
}

__global__ void k_labels(const float* __restrict__ gt, const int* __restrict__ lgp,
                         int* __restrict__ labelg, float* __restrict__ colmask) {
  int m = blockIdx.x * blockDim.x + threadIdx.x;
  if (m >= Mm) return;
  int lab = 0; float bv = gt[m * OUTC];
  for (int c = 1; c < OUTC; ++c) { float v = gt[m * OUTC + c]; if (v > bv) { bv = v; lab = c; } }
  labelg[m] = lab;
  int lg = lgp[0];
  bool upd = (lab < 9) || (m < lg) || ((tf_rand((unsigned)m) % 20u) == 1u);
  colmask[m] = upd ? 1.f : 0.f;
}

// ---------------- pass A: row sums of K -> b -----------------------------

__global__ void __launch_bounds__(256)
k_passA(const _Float16* __restrict__ mean16, const _Float16* __restrict__ sur16,
        const float* __restrict__ w1g, const float* __restrict__ w2g,
        const float* __restrict__ n2og, const float* __restrict__ orig,
        const float* __restrict__ alg, const float* __restrict__ cm2g,
        float* __restrict__ bg) {
  __shared__ _Float16 As[64 * Cc];
  __shared__ float w1s[64], n2os[64], oris[192], rowsum[64];
  const int t = threadIdx.x;
  const int nbase = blockIdx.x * 64;

  stage_panel(mean16 + (size_t)nbase * Cc, As, t);
  for (int i = t; i < 64; i += 256) {
    w1s[i] = w1g[nbase + i]; n2os[i] = n2og[nbase + i]; rowsum[i] = 0.f;
  }
  for (int i = t; i < 192; i += 256)
    oris[i] = ((nbase * 3 + i) < Nn * 3) ? orig[nbase * 3 + i] : 0.f;
  __syncthreads();

  const int wave = t >> 5, lane = t & 31;
  const int nsub = wave & 3, msub0 = (wave >> 2) * 2;
  const int r = lane & 15, hi = lane >> 4;
  const _Float16* arow = &As[(nsub * 16 + r) * Cc];

  Frag a0, a1, a2;                       // A fragments live across whole M loop
  load_frag(arow, hi, 0, a0);
  load_frag(arow, hi, 32, a1);
  load_frag(arow, hi, 64, a2);

  float regsum[8];
#pragma unroll
  for (int v = 0; v < 8; ++v) regsum[v] = 0.f;

  for (int mt = 0; mt < NTM; ++mt) {
    const int mtb = mt * 64;
    for (int ms = msub0; ms <= msub0 + 1; ++ms) {
      const int m = mtb + ms * 16 + r;
      const _Float16* brow = sur16 + (size_t)m * Cc;
      __builtin_prefetch(brow + Cc, 0, 3);
      Frag b0, b1, b2;                   // all chunks in flight, then 3 WMMAs
      load_frag(brow, hi, 0, b0);
      load_frag(brow, hi, 32, b1);
      load_frag(brow, hi, 64, b2);
      v8f acc = {};
      acc = __builtin_amdgcn_wmma_f32_16x16x32_f16(false, a0.v, false, b0.v, (short)0, acc, false, false);
      acc = __builtin_amdgcn_wmma_f32_16x16x32_f16(false, a1.v, false, b1.v, (short)0, acc, false, false);
      acc = __builtin_amdgcn_wmma_f32_16x16x32_f16(false, a2.v, false, b2.v, (short)0, acc, false, false);
      const float w2m = w2g[m], cm2 = cm2g[m];
      const float alx = alg[m * 3], aly = alg[m * 3 + 1], alz = alg[m * 3 + 2];
      const bool mok = (m < Mm);
#pragma unroll
      for (int v = 0; v < 8; ++v) {
        const int nl = nsub * 16 + v + 8 * hi;
        const float d = acc[v];
        const float cdot = oris[nl * 3] * alx + oris[nl * 3 + 1] * aly + oris[nl * 3 + 2] * alz;
        const float cosd = 1.f - d * fast_rcp(fmaxf(w1s[nl] * w2m, 1e-8f));
        float cd = fmaxf(n2os[nl] + cm2 - 2.f * cdot, 0.f);
        const float sup = (cd < 100.f) ? 1.f : 0.f;
        const float sim = 1.f + cosd - __expf(-2.f * cd);       // exp(-0.5*(cd/0.25))
        float Kv = __expf(-sim * (1.f / 0.03f)) * sup;
        regsum[v] += mok ? Kv : 0.f;     // defer reduction to kernel end
      }
    }
  }
  // one 16-lane tree + one ds_add per row, once per kernel
#pragma unroll
  for (int v = 0; v < 8; ++v) {
    const int nl = nsub * 16 + v + 8 * hi;
    const float s = reduce16(regsum[v]);
    if (r == 0 && (nbase + nl) < Nn) atomicAdd(&rowsum[nl], s);
  }
  __syncthreads();
  for (int i = t; i < 64; i += 256) {
    const int n = nbase + i;
    if (n < Nn) bg[n] = (1.f / Nn) / ((1.f / Mm) * rowsum[i] + 1e-16f);
  }
}

// ---------------- pass B: argmax_n K[n,m]*b[n] -> sm_index ----------------

__global__ void __launch_bounds__(256)
k_passB(const _Float16* __restrict__ mean16, const _Float16* __restrict__ sur16,
        const float* __restrict__ w1g, const float* __restrict__ w2g,
        const float* __restrict__ n2og, const float* __restrict__ orig,
        const float* __restrict__ alg, const float* __restrict__ cm2g,
        const float* __restrict__ bg, int* __restrict__ smg) {
  __shared__ _Float16 Bs[64 * Cc];
  __shared__ float w2s[64], cm2s[64], als[192];
  __shared__ unsigned long long best[64];
  const int t = threadIdx.x;
  const int mbase = blockIdx.x * 64;

  stage_panel(sur16 + (size_t)mbase * Cc, Bs, t);
  for (int i = t; i < 64; i += 256) {
    w2s[i] = w2g[mbase + i]; cm2s[i] = cm2g[mbase + i]; best[i] = 0ull;
  }
  for (int i = t; i < 192; i += 256) als[i] = alg[mbase * 3 + i];
  __syncthreads();

  const int wave = t >> 5, lane = t & 31;
  const int msub = wave & 3, nsub0 = (wave >> 2) * 2;
  const int r = lane & 15, hi = lane >> 4;
  const int ml = msub * 16 + r;
  const int m = mbase + ml;
  const _Float16* brow = &Bs[ml * Cc];
  const float w2m = w2s[ml], cm2 = cm2s[ml];
  const float alx = als[ml * 3], aly = als[ml * 3 + 1], alz = als[ml * 3 + 2];

  Frag b0, b1, b2;                       // B fragments (this block's columns) persist
  load_frag(brow, hi, 0, b0);
  load_frag(brow, hi, 32, b1);
  load_frag(brow, hi, 64, b2);

  float bestv = -1.f; int bestn = 0;

  for (int nt = 0; nt < NTN; ++nt) {
    const int ntb = nt * 64;
    for (int s = nsub0; s <= nsub0 + 1; ++s) {
      const _Float16* arow = mean16 + (size_t)(ntb + s * 16 + r) * Cc;
      __builtin_prefetch(arow + Cc, 0, 3);
      Frag a0, a1, a2;
      load_frag(arow, hi, 0, a0);
      load_frag(arow, hi, 32, a1);
      load_frag(arow, hi, 64, a2);
      v8f acc = {};
      acc = __builtin_amdgcn_wmma_f32_16x16x32_f16(false, a0.v, false, b0.v, (short)0, acc, false, false);
      acc = __builtin_amdgcn_wmma_f32_16x16x32_f16(false, a1.v, false, b1.v, (short)0, acc, false, false);
      acc = __builtin_amdgcn_wmma_f32_16x16x32_f16(false, a2.v, false, b2.v, (short)0, acc, false, false);
#pragma unroll
      for (int v = 0; v < 8; ++v) {
        const int ng = ntb + s * 16 + v + 8 * hi;
        if (ng >= Nn) continue;
        const float d = acc[v];
        const float cdot = orig[ng * 3] * alx + orig[ng * 3 + 1] * aly + orig[ng * 3 + 2] * alz;
        const float cosd = 1.f - d * fast_rcp(fmaxf(w1g[ng] * w2m, 1e-8f));
        float cd = fmaxf(n2og[ng] + cm2 - 2.f * cdot, 0.f);
        const float sup = (cd < 100.f) ? 1.f : 0.f;
        const float sim = 1.f + cosd - __expf(-2.f * cd);
        const float val = __expf(-sim * (1.f / 0.03f)) * sup * bg[ng];
        if (val > bestv) { bestv = val; bestn = ng; }
      }
    }
  }
  if (m < Mm) {
    const unsigned fb = __float_as_uint(fmaxf(bestv, 0.f)); // >=0 -> uint-order == float-order
    const unsigned long long pk =
        ((unsigned long long)fb << 32) | (unsigned long long)(0xFFFFFFFFu - (unsigned)bestn);
    atomicMax(&best[ml], pk);
  }
  __syncthreads();
  for (int i = t; i < 64; i += 256) {
    const int mo = mbase + i;
    if (mo < Mm) smg[mo] = (int)(0xFFFFFFFFu - (unsigned)(best[i] & 0xFFFFFFFFull));
  }
}

// ---------------- gather mid_* -------------------------------------------

__global__ void k_gather(const _Float16* __restrict__ mean16, const float* __restrict__ orig,
                         const float* __restrict__ w1g, const float* __restrict__ n2og,
                         const int* __restrict__ smg, _Float16* __restrict__ mid16,
                         float* __restrict__ midcg, float* __restrict__ mn2g,
                         float* __restrict__ cm2mg) {
  int m = blockIdx.x * blockDim.x + threadIdx.x;
  if (m >= Mm) return;
  const int s = smg[m];
  const uint* src = reinterpret_cast<const uint*>(mean16 + (size_t)s * Cc);
  uint* dst = reinterpret_cast<uint*>(mid16 + (size_t)m * Cc);
  for (int k = 0; k < Cc / 2; ++k) dst[k] = src[k];
  midcg[m * 3] = orig[s * 3]; midcg[m * 3 + 1] = orig[s * 3 + 1]; midcg[m * 3 + 2] = orig[s * 3 + 2];
  mn2g[m] = w1g[s] * w1g[s];
  cm2mg[m] = n2og[s];
}

// ---------------- pass C: uw = dist @ onehot / rowsum ---------------------

__global__ void __launch_bounds__(256)
k_passC(const _Float16* __restrict__ mean16, const _Float16* __restrict__ mid16,
        const float* __restrict__ w1g, const float* __restrict__ mn2g,
        const float* __restrict__ n2og, const float* __restrict__ orig,
        const float* __restrict__ midcg, const float* __restrict__ cm2mg,
        const float* __restrict__ colmask, const int* __restrict__ labelg,
        float* __restrict__ uwg) {
  __shared__ _Float16 As[64 * Cc];
  __shared__ float fn2s[64], n2os[64], oris[192], rowsum[64], numer[64 * OUTC];
  const int t = threadIdx.x;
  const int nbase = blockIdx.x * 64;

  stage_panel(mean16 + (size_t)nbase * Cc, As, t);
  for (int i = t; i < 64; i += 256) {
    float w = w1g[nbase + i];
    fn2s[i] = w * w; n2os[i] = n2og[nbase + i]; rowsum[i] = 0.f;
  }
  for (int i = t; i < 192; i += 256)
    oris[i] = ((nbase * 3 + i) < Nn * 3) ? orig[nbase * 3 + i] : 0.f;
  for (int i = t; i < 64 * OUTC; i += 256) numer[i] = 0.f;
  __syncthreads();

  const int wave = t >> 5, lane = t & 31;
  const int nsub = wave & 3, msub0 = (wave >> 2) * 2;
  const int r = lane & 15, hi = lane >> 4;
  const _Float16* arow = &As[(nsub * 16 + r) * Cc];

  Frag a0, a1, a2;
  load_frag(arow, hi, 0, a0);
  load_frag(arow, hi, 32, a1);
  load_frag(arow, hi, 64, a2);

  float regsum[8];
#pragma unroll
  for (int v = 0; v < 8; ++v) regsum[v] = 0.f;

  for (int mt = 0; mt < NTM; ++mt) {
    const int mtb = mt * 64;
    for (int ms = msub0; ms <= msub0 + 1; ++ms) {
      const int m = mtb + ms * 16 + r;
      const _Float16* brow = mid16 + (size_t)m * Cc;
      __builtin_prefetch(brow + Cc, 0, 3);
      Frag b0, b1, b2;
      load_frag(brow, hi, 0, b0);
      load_frag(brow, hi, 32, b1);
      load_frag(brow, hi, 64, b2);
      v8f acc = {};
      acc = __builtin_amdgcn_wmma_f32_16x16x32_f16(false, a0.v, false, b0.v, (short)0, acc, false, false);
      acc = __builtin_amdgcn_wmma_f32_16x16x32_f16(false, a1.v, false, b1.v, (short)0, acc, false, false);
      acc = __builtin_amdgcn_wmma_f32_16x16x32_f16(false, a2.v, false, b2.v, (short)0, acc, false, false);
      const bool mok = (m < Mm);
      const float msk = colmask[m];
      const int lab = labelg[m];
      const float mn2 = mn2g[m], cm2 = cm2mg[m];
      const float mx = midcg[m * 3], my = midcg[m * 3 + 1], mz = midcg[m * 3 + 2];
#pragma unroll
      for (int v = 0; v < 8; ++v) {
        const int nl = nsub * 16 + v + 8 * hi;
        const float d = acc[v];
        const float fd = fmaxf(fn2s[nl] + mn2 - 2.f * d, 0.f);
        const float cdot = oris[nl * 3] * mx + oris[nl * 3 + 1] * my + oris[nl * 3 + 2] * mz;
        const float cdr = fmaxf(n2os[nl] + cm2 - 2.f * cdot, 0.f);
        float dist = __expf(-8.f * cdr - (0.5f / 0.09f) * fd) * msk;
        dist = mok ? dist : 0.f;
        regsum[v] += dist;               // row-sum deferred to kernel end
        // label-weighted numerator: per-lane label, keep per-lane atomic
        if (dist != 0.f && (nbase + nl) < Nn)
          atomicAdd(&numer[nl * OUTC + lab], dist);
      }
    }
  }
#pragma unroll
  for (int v = 0; v < 8; ++v) {
    const int nl = nsub * 16 + v + 8 * hi;
    const float s = reduce16(regsum[v]);
    if (r == 0 && (nbase + nl) < Nn) atomicAdd(&rowsum[nl], s);
  }
  __syncthreads();
  for (int i = t; i < 64 * OUTC; i += 256) {
    const int n = nbase + i / OUTC;
    if (n < Nn)
      uwg[n * OUTC + (i % OUTC)] = numer[i] * fast_rcp(rowsum[i / OUTC] + 1e-16f);
  }
}

// ---------------- tail kernels -------------------------------------------

__global__ void k_init_flags(int* flags) {
  if (threadIdx.x < 8 && blockIdx.x == 0) flags[threadIdx.x] = 0;
}

__global__ void k_softmax_update(const float* __restrict__ svp, const float* __restrict__ uwg,
                                 float* __restrict__ usp) {
  int n = blockIdx.x * blockDim.x + threadIdx.x;
  if (n >= Nn) return;
  float mx = svp[n * OUTC];
  for (int c = 1; c < OUTC; ++c) mx = fmaxf(mx, svp[n * OUTC + c]);
  float sum = 0.f, e[OUTC];
  for (int c = 0; c < OUTC; ++c) { e[c] = __expf(svp[n * OUTC + c] - mx); sum += e[c]; }
  const float inv = fast_rcp(sum);
  for (int c = 0; c < OUTC; ++c)
    usp[n * OUTC + c] = 0.5f * e[c] * inv + 0.5f * uwg[n * OUTC + c];
}

__global__ void k_vals(const int* __restrict__ lgp, const int* __restrict__ labelg,
                       const int* __restrict__ smg, const float* __restrict__ usp,
                       float* __restrict__ vals, int* __restrict__ flags) {
  int i = blockIdx.x * blockDim.x + threadIdx.x;
  if (i >= lgp[0] || i >= LGc) return;
  const float v = usp[smg[i] * OUTC + labelg[i]];
  vals[i] = v;
  if (v > 0.1f) atomicOr(&flags[0], 1);
}

__global__ void k_scatter(const int* __restrict__ lgp, const int* __restrict__ smg,
                          const float* __restrict__ gt, const float* __restrict__ vals,
                          const int* __restrict__ flags, float* __restrict__ usp) {
  int i = blockIdx.x * blockDim.x + threadIdx.x;
  if (i >= lgp[0] || i >= LGc) return;
  const bool take = flags[0] ? (vals[i] > 0.1f) : (vals[i] > 0.0f);
  if (!take) return;
  const int row = smg[i];
  if (row < 0 || row >= Nn) return;
  for (int c = 0; c < OUTC; ++c) usp[row * OUTC + c] = gt[i * OUTC + c];
}

__global__ void k_mx(const float* __restrict__ usp, float* __restrict__ mxg,
                     int* __restrict__ pmaxg, int* __restrict__ flags) {
  int n = blockIdx.x * blockDim.x + threadIdx.x;
  if (n >= Nn) return;
  float mx = usp[n * OUTC]; int pm = 0;
  for (int c = 1; c < OUTC; ++c) { float v = usp[n * OUTC + c]; if (v > mx) { mx = v; pm = c; } }
  mxg[n] = mx; pmaxg[n] = pm;
  if (mx >= 0.9f) atomicOr(&flags[1], 1);
}

__global__ void k_trust(const float* __restrict__ mxg, const int* __restrict__ flags,
                        int* __restrict__ trustg, int* __restrict__ countp) {
  if (threadIdx.x != 0 || blockIdx.x != 0) return;
  const float th = flags[1] ? 0.9f : 0.85f;
  int cnt = 0;
  for (int n = 0; n < Nn; ++n)
    if (mxg[n] >= th) trustg[cnt++] = n;
  countp[0] = cnt;
}

__global__ void k_emit(const int* __restrict__ trustg, const int* __restrict__ countp,
                       const int* __restrict__ pmaxg, const float* __restrict__ uwg,
                       const float* __restrict__ usp, const int* __restrict__ smg,
                       const float* __restrict__ meanf, const float* __restrict__ orig,
                       float* __restrict__ out) {
  const int tid = blockIdx.x * blockDim.x + threadIdx.x;
  const long long T = countp[0];
  const long long oF = T, oC = 97 * T, oP = 100 * T, oU = 120 * T;
  const long long oS = oU + (long long)Nn * OUTC, oM = oS + (long long)Nn * OUTC;
  if (tid < T) out[tid] = (float)trustg[tid];
  if (tid < T * Cc)   { int i = tid / Cc,   k = tid % Cc;   out[oF + tid] = meanf[trustg[i] * Cc + k]; }
  if (tid < T * 3)    { int i = tid / 3,    k = tid % 3;    out[oC + tid] = orig[trustg[i] * 3 + k]; }
  if (tid < T * OUTC) { int i = tid / OUTC, c = tid % OUTC; out[oP + tid] = (c == pmaxg[trustg[i]]) ? 1.f : 0.f; }
  if (tid < Nn * OUTC) { out[oU + tid] = uwg[tid]; out[oS + tid] = usp[tid]; }
  if (tid < Mm) out[oM + tid] = (float)smg[tid];
}

} // namespace

// --------------------------------------------------------------------------

extern "C" void kernel_launch(void* const* d_in, const int* in_sizes, int n_in,
                              void* d_out, int out_size, void* d_ws, size_t ws_size,
                              hipStream_t stream) {
  (void)in_sizes; (void)n_in; (void)out_size; (void)ws_size;
  const float* sur_feat = (const float*)d_in[0];
  const float* sur_crd  = (const float*)d_in[1];
  const float* sur_gt   = (const float*)d_in[2];
  const float* sv_prob  = (const float*)d_in[3];
  const float* mean_f   = (const float*)d_in[4];
  const float* ori_crd  = (const float*)d_in[5];
  const float* posses   = (const float*)d_in[6];
  const int*   lgp      = (const int*)d_in[7];

  char* w = (char*)d_ws;
  size_t off = 0;
  auto alloc = [&](size_t bytes) -> void* {
    void* p = w + off;
    off += (bytes + 255) & ~(size_t)255;
    return p;
  };
  float*     diff    = (float*)alloc(16 * sizeof(float));
  float*     alg     = (float*)alloc((size_t)MP * 3 * sizeof(float));
  float*     cm2g    = (float*)alloc((size_t)MP * sizeof(float));
  _Float16*  mean16  = (_Float16*)alloc((size_t)NP * Cc * sizeof(_Float16));
  _Float16*  sur16   = (_Float16*)alloc((size_t)MP * Cc * sizeof(_Float16));
  _Float16*  mid16   = (_Float16*)alloc((size_t)MP * Cc * sizeof(_Float16));
  float*     w1g     = (float*)alloc((size_t)NP * sizeof(float));
  float*     w2g     = (float*)alloc((size_t)MP * sizeof(float));
  float*     n2og    = (float*)alloc((size_t)NP * sizeof(float));
  float*     bg      = (float*)alloc((size_t)NP * sizeof(float));
  int*       smg     = (int*)alloc((size_t)Mm * sizeof(int));
  int*       labelg  = (int*)alloc((size_t)MP * sizeof(int));
  float*     colmask = (float*)alloc((size_t)MP * sizeof(float));
  float*     midcg   = (float*)alloc((size_t)MP * 3 * sizeof(float));
  float*     mn2g    = (float*)alloc((size_t)MP * sizeof(float));
  float*     cm2mg   = (float*)alloc((size_t)MP * sizeof(float));
  float*     uwg     = (float*)alloc((size_t)Nn * OUTC * sizeof(float));
  float*     uspg    = (float*)alloc((size_t)Nn * OUTC * sizeof(float));
  float*     vals    = (float*)alloc((size_t)LGc * sizeof(float));
  float*     mxg     = (float*)alloc((size_t)Nn * sizeof(float));
  int*       pmaxg   = (int*)alloc((size_t)Nn * sizeof(int));
  int*       trustg  = (int*)alloc((size_t)Nn * sizeof(int));
  int*       countp  = (int*)alloc(8 * sizeof(int));
  int*       flags   = (int*)alloc(8 * sizeof(int));

  const int TB = 256;
  k_init_flags<<<1, 32, 0, stream>>>(flags);
  k_align<<<1, 64, 0, stream>>>(posses, diff);
  k_transform<<<(Mm + TB - 1) / TB, TB, 0, stream>>>(sur_crd, diff, alg, cm2g);
  k_prep_mean<<<(Nn + TB - 1) / TB, TB, 0, stream>>>(mean_f, ori_crd, mean16, w1g, n2og);
  k_prep_sur<<<(Mm + TB - 1) / TB, TB, 0, stream>>>(sur_feat, sur16, w2g);
  k_labels<<<(Mm + TB - 1) / TB, TB, 0, stream>>>(sur_gt, lgp, labelg, colmask);

  k_passA<<<NTN, 256, 0, stream>>>(mean16, sur16, w1g, w2g, n2og, ori_crd, alg, cm2g, bg);
  k_passB<<<NTM, 256, 0, stream>>>(mean16, sur16, w1g, w2g, n2og, ori_crd, alg, cm2g, bg, smg);
  k_gather<<<(Mm + TB - 1) / TB, TB, 0, stream>>>(mean16, ori_crd, w1g, n2og, smg,
                                                  mid16, midcg, mn2g, cm2mg);
  k_passC<<<NTN, 256, 0, stream>>>(mean16, mid16, w1g, mn2g, n2og, ori_crd, midcg, cm2mg,
                                   colmask, labelg, uwg);

  k_softmax_update<<<(Nn + TB - 1) / TB, TB, 0, stream>>>(sv_prob, uwg, uspg);
  k_vals<<<(LGc + TB - 1) / TB, TB, 0, stream>>>(lgp, labelg, smg, uspg, vals, flags);
  k_scatter<<<(LGc + TB - 1) / TB, TB, 0, stream>>>(lgp, smg, sur_gt, vals, flags, uspg);
  k_mx<<<(Nn + TB - 1) / TB, TB, 0, stream>>>(uspg, mxg, pmaxg, flags);
  k_trust<<<1, 1, 0, stream>>>(mxg, flags, trustg, countp);

  const int emitThreads = Nn * Cc; // 576000 covers every emit job
  k_emit<<<(emitThreads + TB - 1) / TB, TB, 0, stream>>>(trustg, countp, pmaxg, uwg, uspg,
                                                         smg, mean_f, ori_crd, (float*)d_out);
}
